// MiniGPT_89197880803333
// MI455X (gfx1250) — matrix operand order repaired and checked by reading, not compiled, not run on hardware
//
#include <hip/hip_runtime.h>
#include <hip/hip_bf16.h>

typedef __attribute__((ext_vector_type(16))) _Float16 v16h;
typedef __attribute__((ext_vector_type(8)))  _Float16 v8h;
typedef __attribute__((ext_vector_type(8)))  float    v8f;
typedef __attribute__((ext_vector_type(4)))  unsigned v4u;
typedef __attribute__((ext_vector_type(8)))  unsigned v8u;

constexpr int kS    = 2048;
constexpr int kD    = 1024;
constexpr int kDff  = 4096;
constexpr int kL    = 6;
constexpr int kH    = 16;
constexpr int kDk   = 64;
constexpr int kV    = 32000;
#define LN_EPS 1e-5f

// Build a v16h A/B fragment from two contiguous 16-byte chunks (global or LDS).
__device__ __forceinline__ v16h ld_frag(const _Float16* lo, const _Float16* hi) {
    v8h a = *(const v8h*)lo;
    v8h b = *(const v8h*)hi;
    v16h r;
#pragma unroll
    for (int i = 0; i < 8; ++i) { r[i] = a[i]; r[8 + i] = b[i]; }
    return r;
}

// ---------------- Tensor Data Mover: 2-D f16 tile global -> LDS ----------------
// Builds D# group0 (count/lds_addr/global_addr/type) and group1 (data_size=2B,
// tensor dims, tile dims, dim0 stride) per CDNA5 ISA 8.3/8.4, then issues
// tensor_load_to_lds (2-SGPR-group form, tensors up to 2D). Tracked by TENSORcnt.
// `lds_escape` is passed as an asm operand so the shared buffer's address escapes
// and the "memory" clobber makes the compiler treat the TDM as writing it
// (otherwise GlobalOpt proves the LDS global is never stored and folds the
// subsequent ds_load fragments to undef).
__device__ __forceinline__ void tdm_load_2d_f16(unsigned lds_byte_off, const _Float16* gptr,
                                                unsigned tensor_d0, unsigned tensor_d1,
                                                unsigned tile_d0, unsigned tile_d1,
                                                unsigned long stride0_elems,
                                                const void* lds_escape) {
    unsigned long ga = (unsigned long)gptr;
    v4u g0;
    g0[0] = 1u;                                        // count=1 valid descriptor
    g0[1] = lds_byte_off;                              // lds_addr [63:32]
    g0[2] = (unsigned)(ga & 0xFFFFFFFFu);              // global_addr[31:0]
    g0[3] = (unsigned)((ga >> 32) & 0x1FFFFFFu)        // global_addr[56:32]
          | (2u << 30);                                // type=2 ("image")
    v8u g1;
    g1[0] = (1u << 16);                                // workgroup_mask=0, data_size=1 (2B)
    g1[1] = (tensor_d0 & 0xFFFFu) << 16;               // tensor_dim0 field @ bits79:48
    g1[2] = (tensor_d0 >> 16) | ((tensor_d1 & 0xFFFFu) << 16);
    g1[3] = (tensor_d1 >> 16) | (tile_d0 << 16);       // tile_dim0 @127:112
    g1[4] = tile_d1;                                   // tile_dim1 @143:128 (tile_dim2=0)
    g1[5] = (unsigned)(stride0_elems & 0xFFFFFFFFu);   // tensor_dim0_stride[31:0]
    g1[6] = (unsigned)((stride0_elems >> 32) & 0xFFFFu);
    g1[7] = 0u;
    asm volatile("tensor_load_to_lds %0, %1"
                 :: "s"(g0), "s"(g1), "v"(lds_escape)
                 : "memory");
}

// ---------------- elementwise / prep kernels ----------------

__global__ void cvt_f32_to_f16_kernel(const float* __restrict__ in,
                                      _Float16* __restrict__ out, size_t n) {
    size_t t = (size_t)blockIdx.x * blockDim.x + threadIdx.x;
    if (t < n) out[t] = (_Float16)in[t];
}

__global__ void embed_kernel(const int* __restrict__ tokens,
                             const float* __restrict__ tok_emb,
                             const float* __restrict__ pos_emb,
                             float* __restrict__ x) {
    size_t t = (size_t)blockIdx.x * blockDim.x + threadIdx.x;   // grid = S*D exactly
    int s = (int)(t / kD);
    int d = (int)(t % kD);
    x[t] = tok_emb[(size_t)tokens[s] * kD + d] + pos_emb[t];
}

// W[K][N] f32 -> Wt[N][K] f16 (coalesced via LDS tile)
__global__ void transpose_cvt_kernel(const float* __restrict__ W,
                                     _Float16* __restrict__ Wt, int K, int N) {
    __shared__ float tile[16][17];
    int k = blockIdx.y * 16 + threadIdx.y;
    int n = blockIdx.x * 16 + threadIdx.x;
    tile[threadIdx.y][threadIdx.x] = W[(size_t)k * N + n];
    __syncthreads();
    int on = blockIdx.x * 16 + threadIdx.y;
    int ok = blockIdx.y * 16 + threadIdx.x;
    Wt[(size_t)on * K + ok] = (_Float16)tile[threadIdx.x][threadIdx.y];
}

// One block (256 threads) per row: h = LN(x)*g + b  -> f16
__global__ __launch_bounds__(256)
void layernorm_kernel(const float* __restrict__ x, const float* __restrict__ g,
                      const float* __restrict__ b, _Float16* __restrict__ out) {
    int row = blockIdx.x;
    int tid = threadIdx.x;
    const float* xr = x + (size_t)row * kD;
    float v[4];
    float s = 0.f, sq = 0.f;
#pragma unroll
    for (int j = 0; j < 4; ++j) {
        v[j] = xr[tid + j * 256];
        s += v[j];
        sq += v[j] * v[j];
    }
#pragma unroll
    for (int m = 16; m >= 1; m >>= 1) {
        s  += __shfl_xor(s, m, 32);
        sq += __shfl_xor(sq, m, 32);
    }
    __shared__ float rs[8], rq[8];
    if ((tid & 31) == 0) { rs[tid >> 5] = s; rq[tid >> 5] = sq; }
    __syncthreads();
    float ts = 0.f, tq = 0.f;
#pragma unroll
    for (int i = 0; i < 8; ++i) { ts += rs[i]; tq += rq[i]; }
    float mu  = ts * (1.0f / kD);
    float var = tq * (1.0f / kD) - mu * mu;
    float rr  = rsqrtf(var + LN_EPS);
#pragma unroll
    for (int j = 0; j < 4; ++j) {
        int d = tid + j * 256;
        out[(size_t)row * kD + d] = (_Float16)((v[j] - mu) * rr * g[d] + b[d]);
    }
}

// ---------------- WMMA GEMM, TDM-fed double-buffered LDS pipeline ----------------
// C[M][N] = A[M][K] (f16 row-major) @ Bt[N][K]^T (f16) + bias
// Block: 8 waves (4 M x 2 N) -> 64x128 output tile.
// Per K-step: TDM stages A slab 64x32 and B slab 128x32 into LDS (double-buffered).
// mode 0: out16[row*N+col] = f16(val)
// mode 1: out16[col*M+row] = f16(val)          (transposed store, for V)
// mode 2: outf[row*N+col] += val               (residual add)
// mode 3: outf[row*N+col]  = val               (fp32 logits)
// act  1: exact GELU
__global__ __launch_bounds__(256)
void gemm_wmma_kernel(const _Float16* __restrict__ A, const _Float16* __restrict__ Bt,
                      const float* __restrict__ bias,
                      _Float16* __restrict__ out16, float* __restrict__ outf,
                      int M, int N, int K, int act, int mode) {
    // Single shared array => allocated at LDS offset 0 (TDM lds_addr is absolute).
    // Layout (bytes): A0 @0 (4KB), A1 @4096, B0 @8192 (8KB), B1 @16384. Total 24KB.
    __shared__ _Float16 sh[12288];

    int lane = threadIdx.x & 31;
    int w    = threadIdx.x >> 5;
    int lr   = lane & 15;
    int half = lane >> 4;
    int r_blk = blockIdx.y * 64;
    int c_blk = blockIdx.x * 128;
    int r0 = r_blk + (w >> 1) * 16;
    int c0 = c_blk + (w & 1) * 64;

    const _Float16* Ag = A  + (size_t)r_blk * K;   // 64 rows
    const _Float16* Bg = Bt + (size_t)c_blk * K;   // 128 rows

    int nk = K / 32;
    if (w == 0) {
        // prologue: fill buffer 0
        tdm_load_2d_f16(0u,    Ag, (unsigned)K, 64u,  32u, 64u,  (unsigned long)K, sh);
        tdm_load_2d_f16(8192u, Bg, (unsigned)K, 128u, 32u, 128u, (unsigned long)K, sh);
    }

    v8f acc[4] = {};

    for (int ik = 0; ik < nk; ++ik) {
        unsigned abuf = (ik & 1) ? 4096u  : 0u;
        unsigned bbuf = (ik & 1) ? 16384u : 8192u;
        __syncthreads();            // everyone done reading the buffer we refill next
        if (w == 0) {
            if (ik + 1 < nk) {
                int kk2 = (ik + 1) * 32;
                unsigned a2 = (ik & 1) ? 0u    : 4096u;
                unsigned b2 = (ik & 1) ? 8192u : 16384u;
                tdm_load_2d_f16(a2, Ag + kk2, (unsigned)(K - kk2), 64u,  32u, 64u,
                                (unsigned long)K, sh);
                tdm_load_2d_f16(b2, Bg + kk2, (unsigned)(K - kk2), 128u, 32u, 128u,
                                (unsigned long)K, sh);
                __builtin_amdgcn_s_wait_tensorcnt(2);   // current slab done, next in flight
            } else {
                __builtin_amdgcn_s_wait_tensorcnt(0);   // last slab done
            }
        }
        __syncthreads();            // current buffer published to all waves

        const _Float16* As = (const _Float16*)((const char*)sh + abuf);
        const _Float16* Bs = (const _Float16*)((const char*)sh + bbuf);
        int ra = (w >> 1) * 16 + lr;                       // local A row
        v16h av = ld_frag(As + ra * 32 + half * 8, As + ra * 32 + half * 8 + 16);
#pragma unroll
        for (int t = 0; t < 4; ++t) {
            int rb = (w & 1) * 64 + t * 16 + lr;           // local B row
            const _Float16* bp = Bs + rb * 32 + half * 8;
            v16h bv = ld_frag(bp, bp + 16);
            acc[t] = __builtin_amdgcn_wmma_f32_16x16x32_f16(
                false, av, false, bv, (short)0, acc[t], false, false);
        }
    }

#pragma unroll
    for (int t = 0; t < 4; ++t) {
#pragma unroll
        for (int i = 0; i < 8; ++i) {
            int row = r0 + i + 8 * half;     // C/D layout: vgpr i -> M=i (+8 for upper half)
            int col = c0 + t * 16 + lr;
            float val = acc[t][i] + (bias ? bias[col] : 0.f);
            if (act == 1) val = 0.5f * val * (1.f + erff(val * 0.70710678118654752f));
            if (mode == 0)      out16[(size_t)row * N + col] = (_Float16)val;
            else if (mode == 1) out16[(size_t)col * M + row] = (_Float16)val;
            else if (mode == 2) outf[(size_t)row * N + col] += val;
            else                outf[(size_t)row * N + col]  = val;
        }
    }
}

// ---------------- flash attention ----------------
// One wave per (16-row q block, head). q,k: [S][D] f16 row-major; vt: [D][S] f16.
__global__ __launch_bounds__(32)
void attn_kernel(const _Float16* __restrict__ q, const _Float16* __restrict__ k,
                 const _Float16* __restrict__ vt, _Float16* __restrict__ ctx) {
    __shared__ _Float16 pl[16 * 32];        // P tile, row-major, 64B rows
    int lane = threadIdx.x;
    int lr = lane & 15, half = lane >> 4;
    int m0 = blockIdx.x * 16;
    int hcol = blockIdx.y * kDk;
    const float scale = 0.125f;             // 1/sqrt(64)

    const _Float16* qr = q + (size_t)(m0 + lr) * kD + hcol;
    v16h aq0 = ld_frag(qr + half * 8,      qr + half * 8 + 16);        // dk 0..31
    v16h aq1 = ld_frag(qr + 32 + half * 8, qr + 32 + half * 8 + 16);   // dk 32..63

    v8f o[4] = {};
    float mrun[8], lrun[8];
#pragma unroll
    for (int i = 0; i < 8; ++i) { mrun[i] = -1e30f; lrun[i] = 0.f; }

    int mlast = m0 + 15;
    for (int n0 = 0; n0 <= mlast; n0 += 32) {
        // scores: q(16x64) @ k^T -> two 16x16 tiles (keys n0..n0+31)
        v8f s[2];
#pragma unroll
        for (int tn = 0; tn < 2; ++tn) {
            const _Float16* kr = k + (size_t)(n0 + tn * 16 + lr) * kD + hcol;
            v16h b0 = ld_frag(kr + half * 8,      kr + half * 8 + 16);
            v16h b1 = ld_frag(kr + 32 + half * 8, kr + 32 + half * 8 + 16);
            v8f z = {};
            z = __builtin_amdgcn_wmma_f32_16x16x32_f16(false, aq0, false, b0, (short)0, z, false, false);
            z = __builtin_amdgcn_wmma_f32_16x16x32_f16(false, aq1, false, b1, (short)0, z, false, false);
            s[tn] = z;
        }
        // online softmax over the 32 new columns
#pragma unroll
        for (int i = 0; i < 8; ++i) {
            int row = m0 + i + 8 * half;
            float v0 = s[0][i] * scale; if (n0 + lr > row)      v0 = -1e30f;
            float v1 = s[1][i] * scale; if (n0 + 16 + lr > row) v1 = -1e30f;
            float rm = fmaxf(v0, v1);
            rm = fmaxf(rm, __shfl_xor(rm, 1, 16));
            rm = fmaxf(rm, __shfl_xor(rm, 2, 16));
            rm = fmaxf(rm, __shfl_xor(rm, 4, 16));
            rm = fmaxf(rm, __shfl_xor(rm, 8, 16));
            float mnew = fmaxf(mrun[i], rm);
            float al   = __expf(mrun[i] - mnew);
            float p0 = __expf(v0 - mnew);
            float p1 = __expf(v1 - mnew);
            float rsum = p0 + p1;
            rsum += __shfl_xor(rsum, 1, 16);
            rsum += __shfl_xor(rsum, 2, 16);
            rsum += __shfl_xor(rsum, 4, 16);
            rsum += __shfl_xor(rsum, 8, 16);
            lrun[i] = lrun[i] * al + rsum;
            mrun[i] = mnew;
#pragma unroll
            for (int t = 0; t < 4; ++t) o[t][i] *= al;
            pl[(i + 8 * half) * 32 + lr]      = (_Float16)p0;
            pl[(i + 8 * half) * 32 + 16 + lr] = (_Float16)p1;
        }
        // O += P(16x32) @ V(32x64); vt rows give contiguous key-dim chunks
        v16h pA = ld_frag(&pl[lr * 32 + half * 8], &pl[lr * 32 + half * 8 + 16]);
#pragma unroll
        for (int t = 0; t < 4; ++t) {
            const _Float16* vr = vt + (size_t)(hcol + t * 16 + lr) * kS + n0 + half * 8;
            v16h bv = ld_frag(vr, vr + 16);
            o[t] = __builtin_amdgcn_wmma_f32_16x16x32_f16(false, pA, false, bv, (short)0, o[t], false, false);
        }
    }
#pragma unroll
    for (int t = 0; t < 4; ++t)
#pragma unroll
        for (int i = 0; i < 8; ++i)
            ctx[(size_t)(m0 + i + 8 * half) * kD + hcol + t * 16 + lr] =
                (_Float16)(o[t][i] / lrun[i]);
}

// ---------------- orchestration ----------------

extern "C" void kernel_launch(void* const* d_in, const int* in_sizes, int n_in,
                              void* d_out, int out_size, void* d_ws, size_t ws_size,
                              hipStream_t stream) {
    (void)in_sizes; (void)n_in; (void)out_size; (void)ws_size;
    const int*   tokens  = (const int*)  d_in[0];
    const float* tok_emb = (const float*)d_in[1];
    const float* pos_emb = (const float*)d_in[2];
    const float* Wq = (const float*)d_in[3],  *bq = (const float*)d_in[4];
    const float* Wk = (const float*)d_in[5],  *bk = (const float*)d_in[6];
    const float* Wv = (const float*)d_in[7],  *bv = (const float*)d_in[8];
    const float* Wo = (const float*)d_in[9],  *bo = (const float*)d_in[10];
    const float* ln1_g = (const float*)d_in[11], *ln1_b = (const float*)d_in[12];
    const float* ln2_g = (const float*)d_in[13], *ln2_b = (const float*)d_in[14];
    const float* W1 = (const float*)d_in[15], *b1 = (const float*)d_in[16];
    const float* W2 = (const float*)d_in[17], *b2 = (const float*)d_in[18];
    const float* lnf_g = (const float*)d_in[19], *lnf_b = (const float*)d_in[20];

    char* ws = (char*)d_ws;
    size_t off = 0;
    auto alloc = [&](size_t bytes) -> void* {
        void* p = ws + off;
        off += (bytes + 255) & ~(size_t)255;
        return p;
    };
    float*    x     = (float*)   alloc((size_t)kS * kD * 4);
    _Float16* h16   = (_Float16*)alloc((size_t)kS * kD * 2);
    _Float16* q16   = (_Float16*)alloc((size_t)kS * kD * 2);
    _Float16* k16   = (_Float16*)alloc((size_t)kS * kD * 2);
    _Float16* vt16  = (_Float16*)alloc((size_t)kS * kD * 2);   // [D][S]
    _Float16* ctx16 = (_Float16*)alloc((size_t)kS * kD * 2);
    _Float16* ffn16 = (_Float16*)alloc((size_t)kS * kDff * 2);
    _Float16* wt    = (_Float16*)alloc((size_t)kDff * kD * 2); // reused transposed weight
    _Float16* emb16 = (_Float16*)alloc((size_t)kV * kD * 2);   // Bt for LM head

    dim3 t16(16, 16);
    dim3 gDD(kD / 16, kD / 16);

    cvt_f32_to_f16_kernel<<<((size_t)kV * kD + 255) / 256, 256, 0, stream>>>(
        tok_emb, emb16, (size_t)kV * kD);
    embed_kernel<<<(kS * kD) / 256, 256, 0, stream>>>(tokens, tok_emb, pos_emb, x);

    for (int l = 0; l < kL; ++l) {
        const float* wq = Wq + (size_t)l * kD * kD;
        const float* wk = Wk + (size_t)l * kD * kD;
        const float* wv = Wv + (size_t)l * kD * kD;
        const float* wo = Wo + (size_t)l * kD * kD;
        const float* w1 = W1 + (size_t)l * kD * kDff;
        const float* w2 = W2 + (size_t)l * kDff * kD;

        // --- attention sublayer ---
        layernorm_kernel<<<kS, 256, 0, stream>>>(x, ln1_g + l * kD, ln1_b + l * kD, h16);

        transpose_cvt_kernel<<<gDD, t16, 0, stream>>>(wq, wt, kD, kD);
        gemm_wmma_kernel<<<dim3(kD / 128, kS / 64), 256, 0, stream>>>(
            h16, wt, bq + l * kD, q16, nullptr, kS, kD, kD, 0, 0);

        transpose_cvt_kernel<<<gDD, t16, 0, stream>>>(wk, wt, kD, kD);
        gemm_wmma_kernel<<<dim3(kD / 128, kS / 64), 256, 0, stream>>>(
            h16, wt, bk + l * kD, k16, nullptr, kS, kD, kD, 0, 0);

        transpose_cvt_kernel<<<gDD, t16, 0, stream>>>(wv, wt, kD, kD);
        gemm_wmma_kernel<<<dim3(kD / 128, kS / 64), 256, 0, stream>>>(
            h16, wt, bv + l * kD, vt16, nullptr, kS, kD, kD, 0, 1);   // transposed store

        attn_kernel<<<dim3(kS / 16, kH), 32, 0, stream>>>(q16, k16, vt16, ctx16);

        transpose_cvt_kernel<<<gDD, t16, 0, stream>>>(wo, wt, kD, kD);
        gemm_wmma_kernel<<<dim3(kD / 128, kS / 64), 256, 0, stream>>>(
            ctx16, wt, bo + l * kD, nullptr, x, kS, kD, kD, 0, 2);    // x += ctx@Wo+bo

        // --- FFN sublayer ---
        layernorm_kernel<<<kS, 256, 0, stream>>>(x, ln2_g + l * kD, ln2_b + l * kD, h16);

        transpose_cvt_kernel<<<dim3(kDff / 16, kD / 16), t16, 0, stream>>>(w1, wt, kD, kDff);
        gemm_wmma_kernel<<<dim3(kDff / 128, kS / 64), 256, 0, stream>>>(
            h16, wt, b1 + l * kDff, ffn16, nullptr, kS, kDff, kD, 1, 0);  // GELU

        transpose_cvt_kernel<<<dim3(kD / 16, kDff / 16), t16, 0, stream>>>(w2, wt, kDff, kD);
        gemm_wmma_kernel<<<dim3(kD / 128, kS / 64), 256, 0, stream>>>(
            ffn16, wt, b2 + l * kD, nullptr, x, kS, kD, kDff, 0, 2);      // x += ffn@W2+b2
    }

    layernorm_kernel<<<kS, 256, 0, stream>>>(x, lnf_g, lnf_b, h16);
    // logits = h @ tok_emb^T  (emb16 is already Bt = [V][K])
    gemm_wmma_kernel<<<dim3(kV / 128, kS / 64), 256, 0, stream>>>(
        h16, emb16, nullptr, nullptr, (float*)d_out, kS, kV, kD, 0, 3);
}
